// FactorizedRandomSelfAttention_3040836845902
// MI455X (gfx1250) — matrix operand order repaired and checked by reading, not compiled, not run on hardware
//
#include <hip/hip_runtime.h>
#include <hip/hip_bf16.h>

typedef __attribute__((ext_vector_type(16))) _Float16     v16h;
typedef __attribute__((ext_vector_type(8)))  float        v8f;
typedef __attribute__((ext_vector_type(4)))  float        f32x4;
typedef __attribute__((ext_vector_type(4)))  unsigned int u32x4;

#define BB 4
#define HH 12
#define NN 2048
#define DD 64
#define KK 32

#define BN 64            // m-tile width per step (softmax update granularity)
#define NMT (NN/BN)      // 32 m-steps
#define NWAVES 8
#define BMWG (16*NWAVES) // 128 rows per workgroup
#define NEGC 1000000.0f

#define R2P 20           // r2s row stride (u32) per m  : 80 B rows, 16B-aligned fragments
#define VSP 36           // vsm row stride (u32) per d  : 144 B rows, 16B-aligned fragments
#define SSP 68           // sS  row stride (f32) per row: 272 B rows, 16B-aligned reads

union F16x16 { v16h h; unsigned int u[8]; u32x4 q[2]; };

__device__ __forceinline__ unsigned int pack2(float x, float y) {
    union { _Float16 h; unsigned short u; } a, b;
    a.h = (_Float16)x; b.h = (_Float16)y;
    return (unsigned int)a.u | ((unsigned int)b.u << 16);
}

__global__ __launch_bounds__(256) void synth_attn_kernel(
    const float* __restrict__ V, const float* __restrict__ mask,
    const float* __restrict__ R1, const float* __restrict__ R2,
    float* __restrict__ X)
{
    // double-buffered tiles
    __shared__ unsigned int r2s[2][BN * R2P];       // 2 x 5 KB   (kpair,m) at [m*R2P+kpair]
    __shared__ unsigned int vsm[2][DD * VSP];       // 2 x 9 KB   (mpair,d) at [d*VSP+mpair]
    __shared__ float sS[NWAVES * 16 * SSP];         // 34 KB, per-wave S scratch
    __shared__ float axm[NWAVES * 16];              // alpha broadcast per row
    __shared__ float lxm[NWAVES * 16];              // 1/l broadcast per row

    const int tid  = threadIdx.x;
    const int lane = tid & 31;
    const int wave = tid >> 5;

    const int ntiles = NN / BMWG; // 16
    const int nt = blockIdx.x % ntiles;
    const int h  = (blockIdx.x / ntiles) % HH;
    const int b  = blockIdx.x / (ntiles * HH);
    const int n0 = nt * BMWG + wave * 16;

    const int lh    = lane & 15;   // 0..15
    const int half  = lane >> 4;   // 0 or 1
    const int kbase = half * 8;

    // fill-role indices (whole block cooperates)
    const int f_m4 = tid & 15, f_kp = tid >> 4;     // R2 group
    const int f_d4 = tid & 15, f_mp = tid >> 4;     // V group 0 (group 1 = +16 mpairs)

    const float* r2base = R2 + (size_t)h * KK * NN;
    const float* vbase  = V + (((size_t)b * HH + h) * NN) * DD;

    // ---- load A-fragment of R1 (16 rows x K=32, f16, contiguous k = 16*half + s) ----
    F16x16 aR1;
    {
        const float* r1p = R1 + ((size_t)h * NN + (n0 + lh)) * KK;
        #pragma unroll
        for (int v = 0; v < 8; ++v) {
            int k0 = 2*kbase + 2*v;
            aR1.u[v] = pack2(r1p[k0], r1p[k0+1]);
        }
    }

    v8f acc[4] = {};            // D=64 -> four 16x16 f32 tiles
    float m_i = -1.0e30f;       // per-lane row stats (row = lh)
    float l_i = 0.0f;

    const float* maskb = mask + (size_t)b * NN;
    float* sSw = &sS[wave * 16 * SSP];

    // prefetch staging registers (packed f16 pairs)
    unsigned int pr2[4], pvv[8];

    // ---- prologue: load + store tile 0 ----
    {
        const float* r2p = r2base;                        // m0 = 0
        f32x4 x0 = *(const f32x4*)(r2p + (size_t)(2*f_kp)   * NN + 4*f_m4);
        f32x4 x1 = *(const f32x4*)(r2p + (size_t)(2*f_kp+1) * NN + 4*f_m4);
        #pragma unroll
        for (int c = 0; c < 4; ++c) r2s[0][(4*f_m4 + c) * R2P + f_kp] = pack2(x0[c], x1[c]);
        const float* vp = vbase;
        #pragma unroll
        for (int i = 0; i < 2; ++i) {
            const int mp = f_mp + 16*i;
            f32x4 y0 = *(const f32x4*)(vp + (size_t)(2*mp)   * DD + 4*f_d4);
            f32x4 y1 = *(const f32x4*)(vp + (size_t)(2*mp+1) * DD + 4*f_d4);
            #pragma unroll
            for (int c = 0; c < 4; ++c) vsm[0][(4*f_d4 + c) * VSP + mp] = pack2(y0[c], y1[c]);
        }
    }
    __syncthreads();

    for (int mt = 0; mt < NMT; ++mt) {
        const int m0  = mt * BN;
        const int cur = mt & 1;
        const unsigned int* r2c = r2s[cur];
        const unsigned int* vsc = vsm[cur];

        // ---- issue global prefetch for tile mt+1 (latency hidden behind compute) ----
        const bool havenext = (mt + 1 < NMT);
        if (havenext) {
            const float* r2p = r2base + (m0 + BN);
            f32x4 x0 = *(const f32x4*)(r2p + (size_t)(2*f_kp)   * NN + 4*f_m4);
            f32x4 x1 = *(const f32x4*)(r2p + (size_t)(2*f_kp+1) * NN + 4*f_m4);
            #pragma unroll
            for (int c = 0; c < 4; ++c) pr2[c] = pack2(x0[c], x1[c]);
            const float* vp = vbase + (size_t)(m0 + BN) * DD;
            #pragma unroll
            for (int i = 0; i < 2; ++i) {
                const int mp = f_mp + 16*i;
                f32x4 y0 = *(const f32x4*)(vp + (size_t)(2*mp)   * DD + 4*f_d4);
                f32x4 y1 = *(const f32x4*)(vp + (size_t)(2*mp+1) * DD + 4*f_d4);
                #pragma unroll
                for (int c = 0; c < 4; ++c) pvv[4*i + c] = pack2(y0[c], y1[c]);
            }
        }

        // ---- S = R1_tile @ R2_tile : four 16x16 tiles; store + mask to per-wave LDS ----
        #pragma unroll
        for (int j = 0; j < 4; ++j) {
            F16x16 bR2;
            const int m = 16*j + lh;
            const u32x4* bp = (const u32x4*)&r2c[m * R2P + kbase];
            bR2.q[0] = bp[0]; bR2.q[1] = bp[1];             // 2x ds_load_b128
            v8f z = {};
            v8f Sj = __builtin_amdgcn_wmma_f32_16x16x32_f16(
                         false, aR1.h, false, bR2.h, (short)0, z, false, false);
            const float madd = -NEGC * (1.0f - maskb[m0 + m]);
            #pragma unroll
            for (int r = 0; r < 8; ++r)
                sSw[(r + half*8) * SSP + 16*j + lh] = Sj[r] + madd;
        }

        // ---- row-wise online softmax, in-lane (row = lh), A-layout column order ----
        float sv[32];
        float alpha;
        {
            const float* srow = &sSw[lh * SSP];
            #pragma unroll
            for (int g = 0; g < 4; ++g) {                   // col groups {0,8,32,40}+2*kbase
                const f32x4* sp = (const f32x4*)(srow + 32*(g>>1) + 2*kbase + 8*(g&1));
                f32x4 a0 = sp[0], a1 = sp[1];
                #pragma unroll
                for (int c = 0; c < 4; ++c) { sv[8*g + c] = a0[c]; sv[8*g + 4 + c] = a1[c]; }
            }
            float rmax = sv[0];
            #pragma unroll
            for (int i = 1; i < 32; ++i) rmax = fmaxf(rmax, sv[i]);
            rmax = fmaxf(rmax, __shfl_xor(rmax, 16, 32));   // combine row's half-lanes

            const float mnew = fmaxf(m_i, rmax);
            alpha = __expf(m_i - mnew);

            float rsum = 0.0f;
            #pragma unroll
            for (int i = 0; i < 32; ++i) { sv[i] = __expf(sv[i] - mnew); rsum += sv[i]; }
            rsum += __shfl_xor(rsum, 16, 32);

            l_i = l_i * alpha + rsum;
            m_i = mnew;
        }

        // broadcast alpha per row (same-wave LDS is in-order; no barrier needed)
        if (lane < 16) axm[wave * 16 + lane] = alpha;

        // ---- pack P A-fragments (sub-tile 0: cols 0..31, 1: cols 32..63) ----
        F16x16 aP[2];
        #pragma unroll
        for (int s = 0; s < 2; ++s)
            #pragma unroll
            for (int v = 0; v < 8; ++v)
                aP[s].u[v] = pack2(sv[16*s + 2*v], sv[16*s + 2*v + 1]);

        // ---- rescale accumulators ----
        float al[8];
        #pragma unroll
        for (int r = 0; r < 8; ++r) al[r] = axm[wave * 16 + half * 8 + r];
        #pragma unroll
        for (int t = 0; t < 4; ++t)
            #pragma unroll
            for (int r = 0; r < 8; ++r)
                acc[t][r] *= al[r];

        // ---- acc += P @ V_tile : K=64 via two chained WMMAs per d-tile (8 total) ----
        #pragma unroll
        for (int t = 0; t < 4; ++t) {
            const int d = 16*t + lh;
            F16x16 bV0, bV1;
            const u32x4* q0 = (const u32x4*)&vsc[d * VSP + kbase];        // mpairs 0..15
            bV0.q[0] = q0[0]; bV0.q[1] = q0[1];
            const u32x4* q1 = (const u32x4*)&vsc[d * VSP + 16 + kbase];   // mpairs 16..31
            bV1.q[0] = q1[0]; bV1.q[1] = q1[1];
            acc[t] = __builtin_amdgcn_wmma_f32_16x16x32_f16(
                         false, aP[0].h, false, bV0.h, (short)0, acc[t], false, false);
            acc[t] = __builtin_amdgcn_wmma_f32_16x16x32_f16(
                         false, aP[1].h, false, bV1.h, (short)0, acc[t], false, false);
        }

        // ---- commit prefetched tile into the other buffer, then one barrier ----
        if (havenext) {
            unsigned int* r2n = r2s[cur ^ 1];
            unsigned int* vsn = vsm[cur ^ 1];
            #pragma unroll
            for (int c = 0; c < 4; ++c) r2n[(4*f_m4 + c) * R2P + f_kp] = pr2[c];
            #pragma unroll
            for (int i = 0; i < 2; ++i) {
                const int mp = f_mp + 16*i;
                #pragma unroll
                for (int c = 0; c < 4; ++c) vsn[(4*f_d4 + c) * VSP + mp] = pvv[4*i + c];
            }
        }
        __syncthreads();
    }

    // ---- epilogue: broadcast 1/l per row, normalize, coalesced f32 stores ----
    if (lane < 16) lxm[wave * 16 + lane] = 1.0f / l_i;
    float li[8];
    #pragma unroll
    for (int r = 0; r < 8; ++r) li[r] = lxm[wave * 16 + half * 8 + r];

    float* xp = X + (((size_t)b * HH + h) * NN) * DD;
    #pragma unroll
    for (int r = 0; r < 8; ++r) {
        const int row = n0 + r + half * 8;
        #pragma unroll
        for (int t = 0; t < 4; ++t)
            xp[(size_t)row * DD + 16*t + lh] = acc[t][r] * li[r];
    }
}

extern "C" void kernel_launch(void* const* d_in, const int* in_sizes, int n_in,
                              void* d_out, int out_size, void* d_ws, size_t ws_size,
                              hipStream_t stream) {
    const float* V    = (const float*)d_in[0];
    const float* mask = (const float*)d_in[1];
    const float* R1   = (const float*)d_in[2];
    const float* R2   = (const float*)d_in[3];
    float* X = (float*)d_out;
    (void)in_sizes; (void)n_in; (void)out_size; (void)d_ws; (void)ws_size;

    const int grid = BB * HH * (NN / BMWG); // 4*12*16 = 768 workgroups
    synth_attn_kernel<<<grid, 256, 0, stream>>>(V, mask, R1, R2, X);
}
